// PointConvSetAbstraction_1013612282420
// MI455X (gfx1250) — matrix operand order repaired and checked by reading, not compiled, not run on hardware
//
#include <hip/hip_runtime.h>

// ---------------- problem constants ----------------
#define BB 8
#define NN 4096
#define DD 61
#define SS 1024
#define KK 32
#define MM (BB * SS * KK)   // 262144 grouped rows
#define GG (BB * SS)        // 8192 groups
#define CEPS 1e-5f

typedef __bf16 v16bf __attribute__((ext_vector_type(16)));
typedef float  v8f   __attribute__((ext_vector_type(8)));

union FragU { v16bf v; float4 f4[2]; };

__device__ __forceinline__ float bf2f(__bf16 x) {
    unsigned short s; __builtin_memcpy(&s, &x, 2);
    return __uint_as_float(((unsigned)s) << 16);
}
__device__ __forceinline__ __bf16 f2bf(float f) {
    unsigned u = __float_as_uint(f);
    u += 0x7FFFu + ((u >> 16) & 1u);          // round-to-nearest-even
    unsigned short s = (unsigned short)(u >> 16);
    __bf16 r; __builtin_memcpy(&r, &s, 2);
    return r;
}
__device__ __forceinline__ v8f vzero8() {
    v8f z = {0.f,0.f,0.f,0.f,0.f,0.f,0.f,0.f};
    return z;
}

// A fragment: 16x32 bf16, row-major source, lda elements.
// lane<16 : elems0-7 = K0..7,  elems8-15 = K16..23   (row = lane)
// lane>=16: elems0-7 = K8..15, elems8-15 = K24..31   (row = lane-16)
__device__ __forceinline__ v16bf load_a16(const __bf16* A, int lda, int lane) {
    int r = lane & 15, h = lane >> 4;
    const __bf16* p = A + r * lda + h * 8;
    FragU u;
    u.f4[0] = *(const float4*)p;
    u.f4[1] = *(const float4*)(p + 16);
    return u.v;
}
// B fragment from weight stored row-major [N][K] (ldb = K stride).
// lane<16 : col = lane,   elems0-7 = K0..7,  elems8-15 = K16..23
// lane>=16: col = lane-16,elems0-7 = K8..15, elems8-15 = K24..31
__device__ __forceinline__ v16bf load_b16(const __bf16* W, int ldb, int lane) {
    int n = lane & 15, h = lane >> 4;
    const __bf16* p = W + n * ldb + h * 8;
    FragU u;
    u.f4[0] = *(const float4*)p;
    u.f4[1] = *(const float4*)(p + 16);
    return u.v;
}

// ---------------- weight conversion ----------------
__global__ void cvt_bf16_kernel(const float* __restrict__ src, __bf16* __restrict__ dst, int n) {
    int i = blockIdx.x * blockDim.x + threadIdx.x;
    if (i < n) dst[i] = f2bf(src[i]);
}

// ---------------- farthest point sampling ----------------
// one workgroup per batch; serial scan of SS steps with LDS dmin + argmax tree
__global__ __launch_bounds__(256) void fps_kernel(const float* __restrict__ xyz, int* __restrict__ fps_idx) {
    const int b = blockIdx.x;
    const int tid = threadIdx.x;
    const float* X = xyz + (size_t)b * 3 * NN;   // [3][N]
    __shared__ float dmin[NN];
    __shared__ float rv[256];
    __shared__ int   ri[256];
    for (int n = tid; n < NN; n += 256) dmin[n] = 1e10f;
    int far = 0;                                  // deterministic start
    if (tid == 0) fps_idx[b * SS] = far;
    __syncthreads();
    for (int t = 1; t < SS; ++t) {
        float px = X[far], py = X[NN + far], pz = X[2 * NN + far];
        float bv = -1e30f; int bi = 0;
        for (int n = tid; n < NN; n += 256) {
            float dx = X[n] - px, dy = X[NN + n] - py, dz = X[2 * NN + n] - pz;
            float d = dx * dx + dy * dy + dz * dz;
            float nd = fminf(dmin[n], d);
            dmin[n] = nd;
            if (nd > bv) { bv = nd; bi = n; }
        }
        rv[tid] = bv; ri[tid] = bi;
        __syncthreads();
        for (int off = 128; off > 0; off >>= 1) {
            if (tid < off) {
                float v2 = rv[tid + off]; int i2 = ri[tid + off];
                if (v2 > rv[tid] || (v2 == rv[tid] && i2 < ri[tid])) { rv[tid] = v2; ri[tid] = i2; }
            }
            __syncthreads();
        }
        far = ri[0];
        if (tid == 0) fps_idx[b * SS + t] = far;
        __syncthreads();
    }
}

// ---------------- kNN (32 smallest, ascending) + new_xyz output ----------------
__global__ __launch_bounds__(256) void knn_kernel(const float* __restrict__ xyz,
                                                  const int* __restrict__ fps_idx,
                                                  int* __restrict__ knn,
                                                  float* __restrict__ out_newxyz) {
    const int bs = blockIdx.x;
    const int b = bs >> 10, s = bs & 1023;
    const int tid = threadIdx.x;
    const float* X = xyz + (size_t)b * 3 * NN;
    const int qi = fps_idx[bs];
    const float qx = X[qi], qy = X[NN + qi], qz = X[2 * NN + qi];
    if (tid < 3) {
        float q = (tid == 0) ? qx : (tid == 1 ? qy : qz);
        out_newxyz[b * 3 * SS + tid * SS + s] = q;
    }
    __shared__ float dist[NN];
    __shared__ float rv[256];
    __shared__ int   ri[256];
    for (int n = tid; n < NN; n += 256) {
        float dx = X[n] - qx, dy = X[NN + n] - qy, dz = X[2 * NN + n] - qz;
        dist[n] = dx * dx + dy * dy + dz * dz;
    }
    __syncthreads();
    for (int j = 0; j < KK; ++j) {
        float bv = 3.3e38f; int bi = 0;
        for (int n = tid; n < NN; n += 256) {
            float d = dist[n];
            if (d < bv) { bv = d; bi = n; }
        }
        rv[tid] = bv; ri[tid] = bi;
        __syncthreads();
        for (int off = 128; off > 0; off >>= 1) {
            if (tid < off) {
                float v2 = rv[tid + off]; int i2 = ri[tid + off];
                if (v2 < rv[tid] || (v2 == rv[tid] && i2 < ri[tid])) { rv[tid] = v2; ri[tid] = i2; }
            }
            __syncthreads();
        }
        if (tid == 0) { knn[bs * KK + j] = ri[0]; dist[ri[0]] = 3.4e38f; }
        __syncthreads();
    }
}

// ---------------- gather grouped features -> bf16 A0 [M,64] + g_norm f32 [M,3] ----------------
__global__ __launch_bounds__(256) void gather_kernel(const float* __restrict__ xyz,
                                                     const float* __restrict__ pts,
                                                     const int* __restrict__ fps_idx,
                                                     const int* __restrict__ knn,
                                                     __bf16* __restrict__ featA,
                                                     float* __restrict__ gnorm) {
    const int bs = blockIdx.x;
    const int b = bs >> 10;
    const float* X = xyz + (size_t)b * 3 * NN;
    const float* P = pts + (size_t)b * DD * NN;
    const int qi = fps_idx[bs];
    for (int e = threadIdx.x; e < KK * 64; e += 256) {
        int k = e >> 6, c = e & 63;
        int n = knn[bs * KK + k];
        long m = (long)bs * KK + k;
        float v;
        if (c < 3) {
            v = X[c * NN + n] - X[c * NN + qi];
            gnorm[m * 3 + c] = v;
        } else {
            v = P[(c - 3) * NN + n];
        }
        featA[m * 64 + c] = f2bf(v);
    }
}

// ---------------- WMMA GEMM: Z[M,COUT] = A[M,CIN] @ W[COUT,CIN]^T ----------------
template<int CIN, int COUT, bool OUT_F32>
__global__ __launch_bounds__(256) void wmma_gemm(const __bf16* __restrict__ A,
                                                 const __bf16* __restrict__ W,
                                                 void* __restrict__ Zout, int M) {
    constexpr int NT = COUT / 16;
    constexpr int KS = CIN / 32;
    const int wave = threadIdx.x >> 5;
    const int lane = threadIdx.x & 31;
    const long m0 = ((long)blockIdx.x * 8 + wave) * 16;
    if (m0 >= M) return;                       // wave-uniform
    v8f acc[NT];
#pragma unroll
    for (int t = 0; t < NT; ++t) acc[t] = vzero8();
    const __bf16* Arow = A + m0 * CIN;
    for (int ks = 0; ks < KS; ++ks) {
        const int kb = ks * 32;
        v16bf a = load_a16(Arow + kb, CIN, lane);
#pragma unroll
        for (int nt = 0; nt < NT; ++nt) {
            v16bf bf = load_b16(W + (long)nt * 16 * CIN + kb, CIN, lane);
            acc[nt] = __builtin_amdgcn_wmma_f32_16x16x32_bf16(
                false, a, false, bf, (short)0, acc[nt], false, false);
        }
    }
    const int n = lane & 15, mh = (lane >> 4) * 8;
#pragma unroll
    for (int nt = 0; nt < NT; ++nt) {
#pragma unroll
        for (int i = 0; i < 8; ++i) {
            long row = m0 + mh + i;
            int col = nt * 16 + n;
            if (OUT_F32) ((float*)Zout)[row * COUT + col] = acc[nt][i];
            else         ((__bf16*)Zout)[row * COUT + col] = f2bf(acc[nt][i]);
        }
    }
}

// ---------------- agg einsum per group: featT[bs](128x32) @ wtsT[bs]^T(32x16) ----------------
__global__ __launch_bounds__(256) void agg_kernel(const __bf16* __restrict__ featT,
                                                  const __bf16* __restrict__ wtsT,
                                                  __bf16* __restrict__ aggA) {
    const int bs = blockIdx.x;
    const int wave = threadIdx.x >> 5;         // m-tile: channels 16*wave..
    const int lane = threadIdx.x & 31;
    const __bf16* Ab = featT + (long)bs * 4096 + wave * 16 * 32;  // [16 rows][32 k]
    const __bf16* Wb = wtsT + (long)bs * 512;                     // [16 n][32 k]
    v16bf a = load_a16(Ab, 32, lane);
    v16bf bfr = load_b16(Wb, 32, lane);
    v8f acc = vzero8();
    acc = __builtin_amdgcn_wmma_f32_16x16x32_bf16(false, a, false, bfr, (short)0, acc, false, false);
    const int n = lane & 15, mh = (lane >> 4) * 8;
#pragma unroll
    for (int i = 0; i < 8; ++i) {
        int row = wave * 16 + mh + i;          // channel c
        aggA[(long)bs * 2048 + row * 16 + n] = f2bf(acc[i]);
    }
}

// ---------------- BN stats: per-channel scale/shift (deterministic tree) ----------------
__global__ __launch_bounds__(256) void bn_stats_bf16(const __bf16* __restrict__ Z,
                                                     const float* __restrict__ g,
                                                     const float* __restrict__ be,
                                                     float* __restrict__ stats, int M, int C) {
    const int c = blockIdx.x, tid = threadIdx.x;
    float s = 0.f, sq = 0.f;
    for (int m = tid; m < M; m += 256) {
        float v = bf2f(Z[(long)m * C + c]);
        s += v; sq += v * v;
    }
    __shared__ float sh[512];
    sh[tid] = s; sh[256 + tid] = sq;
    __syncthreads();
    for (int off = 128; off > 0; off >>= 1) {
        if (tid < off) { sh[tid] += sh[tid + off]; sh[256 + tid] += sh[256 + tid + off]; }
        __syncthreads();
    }
    if (tid == 0) {
        float mean = sh[0] / (float)M;
        float var = sh[256] / (float)M - mean * mean;
        float sc = g[c] * rsqrtf(var + CEPS);
        stats[2 * c] = sc;
        stats[2 * c + 1] = be[c] - mean * sc;
    }
}
__global__ __launch_bounds__(256) void bn_stats_f32(const float* __restrict__ Z,
                                                    const float* __restrict__ g,
                                                    const float* __restrict__ be,
                                                    float* __restrict__ stats, int M, int C) {
    const int c = blockIdx.x, tid = threadIdx.x;
    float s = 0.f, sq = 0.f;
    for (int m = tid; m < M; m += 256) {
        float v = Z[(long)m * C + c];
        s += v; sq += v * v;
    }
    __shared__ float sh[512];
    sh[tid] = s; sh[256 + tid] = sq;
    __syncthreads();
    for (int off = 128; off > 0; off >>= 1) {
        if (tid < off) { sh[tid] += sh[tid + off]; sh[256 + tid] += sh[256 + tid + off]; }
        __syncthreads();
    }
    if (tid == 0) {
        float mean = sh[0] / (float)M;
        float var = sh[256] / (float)M - mean * mean;
        float sc = g[c] * rsqrtf(var + CEPS);
        stats[2 * c] = sc;
        stats[2 * c + 1] = be[c] - mean * sc;
    }
}

// ---------------- BN+ReLU elementwise variants ----------------
__global__ void bn_relu_bf16(const __bf16* __restrict__ Z, const float* __restrict__ st,
                             __bf16* __restrict__ Y, long total, int C) {
    for (long i = (long)blockIdx.x * blockDim.x + threadIdx.x; i < total; i += (long)gridDim.x * blockDim.x) {
        int c = (int)(i % C);
        float y = fmaxf(bf2f(Z[i]) * st[2 * c] + st[2 * c + 1], 0.f);
        Y[i] = f2bf(y);
    }
}
__global__ void bn_relu_f32(const float* __restrict__ Z, const float* __restrict__ st,
                            float* __restrict__ Y, long total, int C) {
    for (long i = (long)blockIdx.x * blockDim.x + threadIdx.x; i < total; i += (long)gridDim.x * blockDim.x) {
        int c = (int)(i % C);
        Y[i] = fmaxf(Z[i] * st[2 * c] + st[2 * c + 1], 0.f);
    }
}
// layer2 BN+ReLU writes group-transposed featT[bs][c][k]
__global__ void bn_relu_featT(const __bf16* __restrict__ Z, const float* __restrict__ st,
                              __bf16* __restrict__ featT) {
    const long total = (long)MM * 128;
    for (long i = (long)blockIdx.x * blockDim.x + threadIdx.x; i < total; i += (long)gridDim.x * blockDim.x) {
        int c = (int)(i & 127);
        long m = i >> 7;
        int k = (int)(m & 31);
        long bs = m >> 5;
        float y = fmaxf(bf2f(Z[i]) * st[2 * c] + st[2 * c + 1], 0.f);
        featT[bs * 4096 + c * 32 + k] = f2bf(y);
    }
}
// weightnet layer2 BN+ReLU writes wtsT[bs][w][k]
__global__ void bn_relu_wtsT(const float* __restrict__ Z, const float* __restrict__ st,
                             __bf16* __restrict__ wtsT) {
    const long total = (long)MM * 16;
    for (long i = (long)blockIdx.x * blockDim.x + threadIdx.x; i < total; i += (long)gridDim.x * blockDim.x) {
        int c = (int)(i & 15);
        long m = i >> 4;
        int k = (int)(m & 31);
        long bs = m >> 5;
        float y = fmaxf(Z[i] * st[2 * c] + st[2 * c + 1], 0.f);
        wtsT[bs * 512 + c * 32 + k] = f2bf(y);
    }
}
// final BN+ReLU -> d_out in [B,128,S] layout
__global__ void bn_relu_out(const float* __restrict__ Z, const float* __restrict__ st,
                            float* __restrict__ out) {
    const long total = (long)GG * 128;
    for (long i = (long)blockIdx.x * blockDim.x + threadIdx.x; i < total; i += (long)gridDim.x * blockDim.x) {
        int c = (int)(i & 127);
        long m = i >> 7;
        int s = (int)(m & 1023);
        int b = (int)(m >> 10);
        float y = fmaxf(Z[i] * st[2 * c] + st[2 * c + 1], 0.f);
        out[(long)b * 131072 + c * 1024 + s] = y;
    }
}

// ---------------- small WeightNet GEMMs (scalar f32) ----------------
template<int CI, int CO>
__global__ __launch_bounds__(256) void wn_gemm(const float* __restrict__ A,
                                               const float* __restrict__ W,
                                               float* __restrict__ Z, int M) {
    int m = blockIdx.x * blockDim.x + threadIdx.x;
    if (m >= M) return;
    float x[CI];
#pragma unroll
    for (int i = 0; i < CI; ++i) x[i] = A[(long)m * CI + i];
#pragma unroll
    for (int o = 0; o < CO; ++o) {
        float acc = 0.f;
#pragma unroll
        for (int i = 0; i < CI; ++i) acc += x[i] * W[o * CI + i];
        Z[(long)m * CO + o] = acc;
    }
}

// ---------------- host orchestration ----------------
extern "C" void kernel_launch(void* const* d_in, const int* in_sizes, int n_in,
                              void* d_out, int out_size, void* d_ws, size_t ws_size,
                              hipStream_t stream) {
    (void)in_sizes; (void)n_in; (void)out_size; (void)ws_size;
    const float* xyz   = (const float*)d_in[0];
    const float* pts   = (const float*)d_in[1];
    const float* mw[3] = {(const float*)d_in[2],  (const float*)d_in[6],  (const float*)d_in[10]};
    const float* mg[3] = {(const float*)d_in[4],  (const float*)d_in[8],  (const float*)d_in[12]};
    const float* mb[3] = {(const float*)d_in[5],  (const float*)d_in[9],  (const float*)d_in[13]};
    const float* ww[3] = {(const float*)d_in[14], (const float*)d_in[18], (const float*)d_in[22]};
    const float* wg[3] = {(const float*)d_in[16], (const float*)d_in[20], (const float*)d_in[24]};
    const float* wb[3] = {(const float*)d_in[17], (const float*)d_in[21], (const float*)d_in[25]};
    const float* lin_w = (const float*)d_in[26];
    const float* bnl_g = (const float*)d_in[28];
    const float* bnl_b = (const float*)d_in[29];

    // workspace carve-up (256B aligned)
    char* ws = (char*)d_ws;
    size_t off = 0;
    auto take = [&](size_t bytes) { char* p = ws + off; off = (off + bytes + 255) & ~(size_t)255; return p; };
    int*    fps   = (int*)take(sizeof(int) * BB * SS);
    int*    knn   = (int*)take(sizeof(int) * (size_t)MM);
    float*  gnorm = (float*)take(sizeof(float) * (size_t)MM * 3);
    __bf16* w0b   = (__bf16*)take(2 * 64 * 64);
    __bf16* w1b   = (__bf16*)take(2 * 64 * 64);
    __bf16* w2b   = (__bf16*)take(2 * 128 * 64);
    __bf16* wlb   = (__bf16*)take(2 * 128 * 2048);
    __bf16* bufA  = (__bf16*)take(2 * (size_t)MM * 128);   // also aliases featT
    __bf16* bufZ  = (__bf16*)take(2 * (size_t)MM * 128);   // also aliases aggA
    float*  wnZ   = (float*)take(sizeof(float) * (size_t)MM * 16);
    float*  wnA   = (float*)take(sizeof(float) * (size_t)MM * 8);
    __bf16* wtsT  = (__bf16*)take(2 * (size_t)GG * 16 * 32);
    float*  zlin  = (float*)take(sizeof(float) * (size_t)GG * 128);
    float*  stats = (float*)take(sizeof(float) * 256 * 2);
    __bf16* featT = bufA;
    __bf16* aggA  = bufZ;

    // weight conversion to bf16
    cvt_bf16_kernel<<<(64 * 64 + 255) / 256, 256, 0, stream>>>(mw[0], w0b, 64 * 64);
    cvt_bf16_kernel<<<(64 * 64 + 255) / 256, 256, 0, stream>>>(mw[1], w1b, 64 * 64);
    cvt_bf16_kernel<<<(128 * 64 + 255) / 256, 256, 0, stream>>>(mw[2], w2b, 128 * 64);
    cvt_bf16_kernel<<<(128 * 2048 + 255) / 256, 256, 0, stream>>>(lin_w, wlb, 128 * 2048);

    // sampling / grouping
    fps_kernel<<<BB, 256, 0, stream>>>(xyz, fps);
    knn_kernel<<<GG, 256, 0, stream>>>(xyz, fps, knn, (float*)d_out);
    gather_kernel<<<GG, 256, 0, stream>>>(xyz, pts, fps, knn, bufA, gnorm);

    const unsigned gemm_blocks = MM / (16 * 8);   // 2048
    const long e64 = (long)MM * 64, e128 = (long)MM * 128;

    // MLP layer 0: 64 -> 64
    wmma_gemm<64, 64, false><<<gemm_blocks, 256, 0, stream>>>(bufA, w0b, bufZ, MM);
    bn_stats_bf16<<<64, 256, 0, stream>>>(bufZ, mg[0], mb[0], stats, MM, 64);
    bn_relu_bf16<<<(unsigned)((e64 + 255) / 256), 256, 0, stream>>>(bufZ, stats, bufA, e64, 64);
    // MLP layer 1: 64 -> 64
    wmma_gemm<64, 64, false><<<gemm_blocks, 256, 0, stream>>>(bufA, w1b, bufZ, MM);
    bn_stats_bf16<<<64, 256, 0, stream>>>(bufZ, mg[1], mb[1], stats, MM, 64);
    bn_relu_bf16<<<(unsigned)((e64 + 255) / 256), 256, 0, stream>>>(bufZ, stats, bufA, e64, 64);
    // MLP layer 2: 64 -> 128, BN+ReLU into group-transposed featT
    wmma_gemm<64, 128, false><<<gemm_blocks, 256, 0, stream>>>(bufA, w2b, bufZ, MM);
    bn_stats_bf16<<<128, 256, 0, stream>>>(bufZ, mg[2], mb[2], stats, MM, 128);
    bn_relu_featT<<<(unsigned)((e128 + 255) / 256), 256, 0, stream>>>(bufZ, stats, featT);

    // WeightNet: 3 -> 8 -> 8 -> 16 (scalar), BN between layers
    wn_gemm<3, 8><<<(MM + 255) / 256, 256, 0, stream>>>(gnorm, ww[0], wnZ, MM);
    bn_stats_f32<<<8, 256, 0, stream>>>(wnZ, wg[0], wb[0], stats, MM, 8);
    bn_relu_f32<<<(unsigned)(((long)MM * 8 + 255) / 256), 256, 0, stream>>>(wnZ, stats, wnA, (long)MM * 8, 8);
    wn_gemm<8, 8><<<(MM + 255) / 256, 256, 0, stream>>>(wnA, ww[1], wnZ, MM);
    bn_stats_f32<<<8, 256, 0, stream>>>(wnZ, wg[1], wb[1], stats, MM, 8);
    bn_relu_f32<<<(unsigned)(((long)MM * 8 + 255) / 256), 256, 0, stream>>>(wnZ, stats, wnA, (long)MM * 8, 8);
    wn_gemm<8, 16><<<(MM + 255) / 256, 256, 0, stream>>>(wnA, ww[2], wnZ, MM);
    bn_stats_f32<<<16, 256, 0, stream>>>(wnZ, wg[2], wb[2], stats, MM, 16);
    bn_relu_wtsT<<<(unsigned)(((long)MM * 16 + 255) / 256), 256, 0, stream>>>(wnZ, stats, wtsT);

    // agg einsum (per-group 128x32 @ 32x16 WMMA) -> aggA [G, 2048] bf16
    agg_kernel<<<GG, 256, 0, stream>>>(featT, wtsT, aggA);

    // Linear [8192,2048] @ [2048,128]^T -> zlin f32
    wmma_gemm<2048, 128, true><<<GG / (16 * 8), 256, 0, stream>>>(aggA, wlb, zlin, GG);

    // final BN (over B,S) + ReLU -> d_out[B,128,S] (after the 24576-float new_xyz block)
    bn_stats_f32<<<128, 256, 0, stream>>>(zlin, bnl_g, bnl_b, stats, GG, 128);
    bn_relu_out<<<(unsigned)(((long)GG * 128 + 255) / 256), 256, 0, stream>>>(
        zlin, stats, (float*)d_out + BB * 3 * SS);
}
